// CrossAttention_67242007986924
// MI455X (gfx1250) — matrix operand order repaired and checked by reading, not compiled, not run on hardware
//
#include <hip/hip_runtime.h>

typedef __attribute__((ext_vector_type(16))) __bf16 v16bf;
typedef __attribute__((ext_vector_type(8)))  __bf16 v8bf;
typedef __attribute__((ext_vector_type(8)))  float  v8f;

#define DIMD 512
#define NCTX 4096

__device__ __forceinline__ __bf16 f2bf(float f) {
  union { float f; unsigned u; } v; v.f = f;
  unsigned r = v.u + 0x7FFFu + ((v.u >> 16) & 1u);   // round-to-nearest-even
  unsigned short h = (unsigned short)(r >> 16);
  return __builtin_bit_cast(__bf16, h);
}

__device__ __forceinline__ v8f wmma_bf16(v16bf a, v16bf b, v8f c) {
  // D(f32 16x16) = A(bf16 16x32) * B(bf16 32x16) + C
  return __builtin_amdgcn_wmma_f32_16x16x32_bf16(false, a, false, b, (short)0, c,
                                                 false, false);
}

__device__ __forceinline__ v16bf pack16(v8bf lo, v8bf hi) {
  v16bf r;
#pragma unroll
  for (int j = 0; j < 8; ++j) { r[j] = lo[j]; r[8 + j] = hi[j]; }
  return r;
}

// async global->LDS 16B copy (gfx1250 GV mode: VDST = LDS byte addr, VADDR = 64b)
__device__ __forceinline__ void async_copy_b128(unsigned lds_addr, const void* gp) {
  asm volatile("global_load_async_to_lds_b128 %0, %1, off"
               :: "v"(lds_addr), "v"((unsigned long long)gp) : "memory");
}
__device__ __forceinline__ void wait_async0() {
  asm volatile("s_wait_asynccnt 0x0" ::: "memory");
}

// ---------------------------------------------------------------- cast f32->bf16
__global__ void ca_cast_kernel(const float* __restrict__ s, __bf16* __restrict__ d,
                               long n) {
  long i = (long)blockIdx.x * blockDim.x + threadIdx.x;
  long stride = (long)gridDim.x * blockDim.x;
  for (; i < n; i += stride) d[i] = f2bf(s[i]);
}

// ---------------------------------------------------------------- weight cast + transpose
// W[K,N] f32 row-major -> Wt[N,K] bf16 row-major
__global__ __launch_bounds__(256) void ca_wtrans_kernel(
    const float* __restrict__ W, __bf16* __restrict__ Wt, int K, int N) {
  int k = blockIdx.x;                      // 0..K-1
  for (int n = threadIdx.x; n < N; n += 256)
    Wt[(size_t)n * K + k] = f2bf(W[(size_t)k * N + n]);
}

// ---------------------------------------------------------------- WMMA GEMM
// C[M,N] (f32) = A[M,K] (bf16, row-major) * Wt[N,K] (bf16, row-major) (+ bias[N])
// block tile 64x64, 4 waves (2x2), each wave 32x32, K-step 32.
// Wt tile (64 cols x 32 k) staged to LDS via async global->LDS copies.
__global__ __launch_bounds__(128) void ca_gemm_kernel(
    const __bf16* __restrict__ A, const __bf16* __restrict__ Wt,
    float* __restrict__ C, int M, int N, int K, const float* __restrict__ bias) {
  __shared__ __bf16 Bs[64][40];            // [col][k], 80B row pitch (padded)

  const int tid  = threadIdx.x;
  const int wv   = tid >> 5;
  const int lane = tid & 31;
  const int lh   = lane & 15;
  const int hi   = lane >> 4;
  const int row0 = blockIdx.y * 64;
  const int col0 = blockIdx.x * 64;
  const int wm   = wv >> 1;
  const int wn   = wv & 1;

  const unsigned lds_base = (unsigned)(unsigned long long)(void*)&Bs[0][0];

  v8f acc[2][2];
#pragma unroll
  for (int mt = 0; mt < 2; ++mt)
#pragma unroll
    for (int nt = 0; nt < 2; ++nt)
#pragma unroll
      for (int r = 0; r < 8; ++r) acc[mt][nt][r] = 0.f;

  for (int k0 = 0; k0 < K; k0 += 32) {
    // async-stage Wt tile: 64 rows x 32 k (64B/row) = 256 x 16B chunks
#pragma unroll
    for (int i = 0; i < 2; ++i) {
      int cid = tid * 2 + i;               // 0..255
      int r   = cid >> 2;                  // tile row (output col), 0..63
      int c   = cid & 3;                   // 16B chunk within row, 0..3
      const __bf16* g = Wt + (size_t)(col0 + r) * K + k0 + c * 8;
      async_copy_b128(lds_base + (unsigned)(r * 80 + c * 16), g);
    }
    wait_async0();
    __syncthreads();

    v16bf af[2], bfr[2];
#pragma unroll
    for (int mt = 0; mt < 2; ++mt) {
      const __bf16* p = A + (size_t)(row0 + wm * 32 + mt * 16 + lh) * K + k0 + hi * 8;
      af[mt] = pack16(*(const v8bf*)p, *(const v8bf*)(p + 16));
    }
#pragma unroll
    for (int nt = 0; nt < 2; ++nt) {
      const __bf16* p = &Bs[wn * 32 + nt * 16 + lh][hi * 8];
      bfr[nt] = pack16(*(const v8bf*)p, *(const v8bf*)(p + 16));
    }
#pragma unroll
    for (int mt = 0; mt < 2; ++mt)
#pragma unroll
      for (int nt = 0; nt < 2; ++nt)
        acc[mt][nt] = wmma_bf16(af[mt], bfr[nt], acc[mt][nt]);
    __syncthreads();                        // protect Bs before next stage
  }

#pragma unroll
  for (int mt = 0; mt < 2; ++mt)
#pragma unroll
    for (int nt = 0; nt < 2; ++nt) {
      int col = col0 + wn * 32 + nt * 16 + lh;
      float bb = bias ? bias[col] : 0.f;
#pragma unroll
      for (int r = 0; r < 8; ++r) {
        int row = row0 + wm * 32 + mt * 16 + r + 8 * hi;
        C[(size_t)row * N + col] = acc[mt][nt][r] + bb;
      }
    }
}

// ---------------------------------------------------------------- LayerNorm + cast
__global__ __launch_bounds__(256) void ca_ln_kernel(
    const float* __restrict__ src, int ld, const float* __restrict__ g,
    const float* __restrict__ bta, __bf16* __restrict__ dst) {
  __shared__ float ssum[256], ssq[256];
  const int row = blockIdx.x, t = threadIdx.x;
  const float* p = src + (size_t)row * ld;
  float x0 = p[t], x1 = p[t + 256];
  ssum[t] = x0 + x1;
  ssq[t]  = x0 * x0 + x1 * x1;
  __syncthreads();
  for (int o = 128; o > 0; o >>= 1) {
    if (t < o) { ssum[t] += ssum[t + o]; ssq[t] += ssq[t + o]; }
    __syncthreads();
  }
  float mu  = ssum[0] * (1.f / 512.f);
  float var = ssq[0] * (1.f / 512.f) - mu * mu;
  float rs  = rsqrtf(var + 1e-5f);
  dst[(size_t)row * 512 + t]       = f2bf((x0 - mu) * rs * g[t] + bta[t]);
  dst[(size_t)row * 512 + t + 256] = f2bf((x1 - mu) * rs * g[t + 256] + bta[t + 256]);
}

// ---------------------------------------------------------------- V cast + transpose
__global__ __launch_bounds__(256) void ca_vtrans_kernel(
    const float* __restrict__ kv, __bf16* __restrict__ vt) {
  const int row = blockIdx.x;              // 0..16383
  const int b = row >> 12, mr = row & 4095;
  const int t = threadIdx.x;
  const float* p = kv + (size_t)row * 1024 + 512;
  __bf16* o = vt + (size_t)b * DIMD * NCTX;
  o[(size_t)t * NCTX + mr]         = f2bf(p[t]);
  o[(size_t)(t + 256) * NCTX + mr] = f2bf(p[t + 256]);
}

// ---------------------------------------------------------------- flash attention
// block: 4 waves, 16 queries. Wave w owns d-slice [w*128, w*128+128).
// S^T = K * Q^T (per-wave partials reduced via LDS), online softmax in registers,
// O^T = V^T * P^T.  SCALE = 64^-0.5 = 0.125
__global__ __launch_bounds__(128) void ca_flash_kernel(
    const __bf16* __restrict__ Q, const __bf16* __restrict__ K,
    const __bf16* __restrict__ V /* [b][d][m] */, __bf16* __restrict__ O) {
  __shared__ float red[4][2][32][8];       // 8KB: [wave][ktile][lane][vgpr]

  const int q0 = blockIdx.x * 16;
  const int b  = blockIdx.y;
  const int tid = threadIdx.x;
  const int wv   = tid >> 5;
  const int lane = tid & 31;
  const int lh   = lane & 15;
  const int hi   = lane >> 4;

  const __bf16* Qb = Q + (size_t)b * NCTX * DIMD;
  const __bf16* Kb = K + (size_t)b * NCTX * DIMD;
  const __bf16* Vb = V + (size_t)b * DIMD * NCTX;

  v16bf qf[4];
#pragma unroll
  for (int s = 0; s < 4; ++s) {
    const __bf16* p = Qb + (size_t)(q0 + lh) * DIMD + wv * 128 + s * 32 + hi * 16;
    qf[s] = *(const v16bf*)p;
  }

  v8f acc[8];
#pragma unroll
  for (int t = 0; t < 8; ++t)
#pragma unroll
    for (int r = 0; r < 8; ++r) acc[t][r] = 0.f;

  float m_run = -1e30f, l_run = 0.f;
  const float SC = 0.125f;

  for (int k0 = 0; k0 < NCTX; k0 += 32) {
    if (k0 + 32 < NCTX)
      __builtin_prefetch(Kb + (size_t)(k0 + 32 + lh) * DIMD + wv * 128, 0, 1);

    v8f st[2];
#pragma unroll
    for (int kt = 0; kt < 2; ++kt) {
      v8f sa;
#pragma unroll
      for (int r = 0; r < 8; ++r) sa[r] = 0.f;
#pragma unroll
      for (int s = 0; s < 4; ++s) {
        const __bf16* p =
            Kb + (size_t)(k0 + kt * 16 + lh) * DIMD + wv * 128 + s * 32 + hi * 16;
        v16bf af = pack16(*(const v8bf*)p, *(const v8bf*)(p + 8));
        sa = wmma_bf16(af, qf[s], sa);
      }
      st[kt] = sa;
    }

#pragma unroll
    for (int kt = 0; kt < 2; ++kt)
#pragma unroll
      for (int j = 0; j < 8; ++j) red[wv][kt][lane][j] = st[kt][j];
    __syncthreads();
    float t0[8], t1[8];
#pragma unroll
    for (int j = 0; j < 8; ++j) {
      t0[j] = (red[0][0][lane][j] + red[1][0][lane][j] +
               red[2][0][lane][j] + red[3][0][lane][j]) * SC;
      t1[j] = (red[0][1][lane][j] + red[1][1][lane][j] +
               red[2][1][lane][j] + red[3][1][lane][j]) * SC;
    }
    __syncthreads();

    float rmax = -1e30f;
#pragma unroll
    for (int j = 0; j < 8; ++j) rmax = fmaxf(rmax, fmaxf(t0[j], t1[j]));
    rmax = fmaxf(rmax, __shfl_xor(rmax, 16, 32));
    float m_new = fmaxf(m_run, rmax);
    float alpha = __expf(m_run - m_new);
    float rsum = 0.f;
#pragma unroll
    for (int j = 0; j < 8; ++j) {
      t0[j] = __expf(t0[j] - m_new);
      t1[j] = __expf(t1[j] - m_new);
      rsum += t0[j] + t1[j];
    }
    rsum += __shfl_xor(rsum, 16, 32);
    l_run = l_run * alpha + rsum;
    m_run = m_new;

#pragma unroll
    for (int t = 0; t < 8; ++t)
#pragma unroll
      for (int r = 0; r < 8; ++r) acc[t][r] *= alpha;

    v16bf pf;
#pragma unroll
    for (int j = 0; j < 8; ++j) {
      float o0 = __shfl_xor(t0[j], 16, 32);
      float o1 = __shfl_xor(t1[j], 16, 32);
      pf[j]     = f2bf(hi ? o1 : t0[j]);
      pf[8 + j] = f2bf(hi ? t1[j] : o0);
    }

#pragma unroll
    for (int dt = 0; dt < 8; ++dt) {
      const __bf16* p = Vb + (size_t)(wv * 128 + dt * 16 + lh) * NCTX + k0 + hi * 16;
      v16bf vf = pack16(*(const v8bf*)p, *(const v8bf*)(p + 8));
      acc[dt] = wmma_bf16(vf, pf, acc[dt]);
    }
  }

  float inv = 1.0f / l_run;
#pragma unroll
  for (int dt = 0; dt < 8; ++dt)
#pragma unroll
    for (int r = 0; r < 8; ++r) {
      int d = wv * 128 + dt * 16 + r + 8 * hi;
      O[((size_t)b * NCTX + q0 + lh) * DIMD + d] = f2bf(acc[dt][r] * inv);
    }
}

// ---------------------------------------------------------------- launch
extern "C" void kernel_launch(void* const* d_in, const int* in_sizes, int n_in,
                              void* d_out, int out_size, void* d_ws, size_t ws_size,
                              hipStream_t stream) {
  (void)in_sizes; (void)n_in; (void)out_size; (void)ws_size;
  const float* x   = (const float*)d_in[0];
  const float* ctx = (const float*)d_in[1];
  const float* Wq  = (const float*)d_in[2];
  const float* Wkv = (const float*)d_in[3];
  const float* Wo  = (const float*)d_in[4];
  const float* bo  = (const float*)d_in[5];
  const float* lng = (const float*)d_in[6];
  const float* lnb = (const float*)d_in[7];
  float* out = (float*)d_out;

  const size_t MB = 1ull << 20;
  char* w = (char*)d_ws;
  __bf16* xh   = (__bf16*)(w + 0);        // 16MB (reused as attn_out later)
  __bf16* chx  = (__bf16*)(w + 16 * MB);  // 16MB
  __bf16* Wqt  = (__bf16*)(w + 32 * MB);  // 0.5MB  Wt[512][512]
  __bf16* Wkvt = (__bf16*)(w + 33 * MB);  // 1MB    Wt[1024][512]
  __bf16* Wot  = (__bf16*)(w + 34 * MB);  // 0.5MB  Wt[512][512]
  __bf16* Qh   = (__bf16*)(w + 35 * MB);  // 16MB
  __bf16* Kh   = (__bf16*)(w + 51 * MB);  // 16MB
  __bf16* Vt   = (__bf16*)(w + 67 * MB);  // 16MB
  float*  Qf   = (float*)(w + 83 * MB);   // 32MB
  float*  KVf  = (float*)(w + 115 * MB);  // 64MB (end 179MB)
  __bf16* Ah   = xh;                      // attn output overlays dead xh

  // activations cast; weights cast+transpose (tiny, L2-resident)
  ca_cast_kernel<<<4096, 256, 0, stream>>>(x,   xh,  16384L * 512);
  ca_cast_kernel<<<4096, 256, 0, stream>>>(ctx, chx, 16384L * 512);
  ca_wtrans_kernel<<<512, 256, 0, stream>>>(Wq,  Wqt,  512, 512);
  ca_wtrans_kernel<<<512, 256, 0, stream>>>(Wkv, Wkvt, 512, 1024);
  ca_wtrans_kernel<<<512, 256, 0, stream>>>(Wo,  Wot,  512, 512);

  // projections
  ca_gemm_kernel<<<dim3(8, 256),  128, 0, stream>>>(xh,  Wqt,  Qf,  16384, 512,  512, nullptr);
  ca_gemm_kernel<<<dim3(16, 256), 128, 0, stream>>>(chx, Wkvt, KVf, 16384, 1024, 512, nullptr);

  // qk-norm + casts (+ V transpose)
  ca_ln_kernel<<<16384, 256, 0, stream>>>(Qf,  512,  lng, lnb, Qh);
  ca_ln_kernel<<<16384, 256, 0, stream>>>(KVf, 1024, lng, lnb, Kh);
  ca_vtrans_kernel<<<16384, 256, 0, stream>>>(KVf, Vt);

  // fused attention
  ca_flash_kernel<<<dim3(256, 4), 128, 0, stream>>>(Qh, Kh, Vt, Ah);

  // output projection + bias
  ca_gemm_kernel<<<dim3(8, 256), 128, 0, stream>>>(Ah, Wot, out, 16384, 512, 512, bo);
}
